// CUDABlockMaskAttention_9363028705734
// MI455X (gfx1250) — compile-verified, matrix operand order checked by reading, not run on hardware
//
#include <hip/hip_runtime.h>
#include <hip/hip_bf16.h>

// Shapes fixed by the reference setup.
#define BATCH 2
#define HEADS 8
#define SEQ   2048
#define DIM   64
#define BLK   64
#define NB    (SEQ / BLK)   // 32
#define LDSS  72            // padded row stride in halves (144B, 16B-aligned)

typedef __attribute__((ext_vector_type(16))) _Float16 v16h;
typedef __attribute__((ext_vector_type(8)))  _Float16 v8h;
typedef __attribute__((ext_vector_type(8)))  float    v8f;

union V16U { v16h v; v8h h[2]; };

// Load 8 consecutive f32 from global and convert to 8 f16 (compiler emits b128 loads).
static __device__ __forceinline__ v8h ld8_cvt_g(const float* __restrict__ p) {
    v8h r;
#pragma unroll
    for (int i = 0; i < 8; ++i) r[i] = (_Float16)p[i];
    return r;
}

// Assemble a 16-bit WMMA operand fragment from a row-contiguous f16 array:
// chunk0 at element offset k0, chunk1 at k0+16 (ISA 16-bit A/B operand layout).
static __device__ __forceinline__ v16h ld_frag_lds(const _Float16* __restrict__ row, int k0) {
    V16U u;
    u.h[0] = *(const v8h*)(row + k0);
    u.h[1] = *(const v8h*)(row + k0 + 16);
    return u.v;
}

__global__ void __launch_bounds__(128)
block_mask_attn_kernel(const float* __restrict__ Q,
                       const float* __restrict__ K,
                       const float* __restrict__ V,
                       const float* __restrict__ MASK,  // [B, S]
                       const int*   __restrict__ MAT,   // [NB, NB]
                       float* __restrict__ OUT) {
    __shared__ __align__(16) _Float16 sK[BLK * LDSS];        // [key][d]
    __shared__ __align__(16) _Float16 sV[DIM * LDSS];        // transposed: [d][key]
    __shared__ __align__(16) _Float16 sP[4 * 16 * LDSS];     // per-wave P tiles [row][key]

    const int qb   = blockIdx.x;   // query block 0..31
    const int h    = blockIdx.y;
    const int b    = blockIdx.z;
    const int tid  = threadIdx.x;  // 0..127
    const int wave = tid >> 5;     // 0..3  (16-row strip)
    const int lane = tid & 31;
    const int l15  = lane & 15;
    const int c0   = (lane & 16) ? 8 : 0;   // K-offset of chunk0 per ISA operand layout

    const long  bh      = (long)(b * HEADS + h);
    const float* Qbh    = Q + bh * SEQ * DIM;
    const float* Kbh    = K + bh * SEQ * DIM;
    const float* Vbh    = V + bh * SEQ * DIM;
    float*       Obh    = OUT + bh * SEQ * DIM;
    const float* mrow   = MASK + (long)b * SEQ;

    // ---- Q A-fragments for this wave's 16 rows, resident for whole kernel ----
    const int qrow = qb * BLK + wave * 16 + l15;
    const float* qp = Qbh + (long)qrow * DIM;
    V16U q0u, q1u;
    q0u.h[0] = ld8_cvt_g(qp + c0);           // K = c0..c0+7
    q0u.h[1] = ld8_cvt_g(qp + 16 + c0);      // K = 16+c0..
    q1u.h[0] = ld8_cvt_g(qp + 32 + c0);
    q1u.h[1] = ld8_cvt_g(qp + 48 + c0);
    const v16h qf0 = q0u.v, qf1 = q1u.v;

    // ---- flash-attention running state (per lane: 8 rows' worth, see C/D layout) ----
    float mrun[8], lrun[8];
    v8f   acc[4];
#pragma unroll
    for (int r = 0; r < 8; ++r) { mrun[r] = -1e30f; lrun[r] = 0.0f; }
#pragma unroll
    for (int ot = 0; ot < 4; ++ot) acc[ot] = (v8f)0.0f;

    // staging assignment: thread -> (row, 32-wide half of D)
    const int srow  = tid >> 1;
    const int shalf = (tid & 1) * 32;

    _Float16* pbase = sP + (wave * 16 + l15) * LDSS;   // this lane's P row

    for (int kb = 0; kb < NB; ++kb) {
        if (MAT[qb * NB + kb] == 0) continue;          // uniform across workgroup

        __syncthreads();  // previous iteration done reading sK/sV

        // ---- stage K[key][d] and V^T[d][key] as f16 into LDS ----
        const float* gk = Kbh + (long)(kb * BLK + srow) * DIM + shalf;
        const float* gv = Vbh + (long)(kb * BLK + srow) * DIM + shalf;
        __builtin_prefetch(gk + BLK * DIM, 0, 1);      // pull next key block toward L2
        __builtin_prefetch(gv + BLK * DIM, 0, 1);
#pragma unroll
        for (int i = 0; i < 32; i += 8) {
            v8h kk = ld8_cvt_g(gk + i);
            *(v8h*)(sK + srow * LDSS + shalf + i) = kk;
            v8h vv = ld8_cvt_g(gv + i);
#pragma unroll
            for (int j = 0; j < 8; ++j)
                sV[(shalf + i + j) * LDSS + srow] = vv[j];   // transpose into [d][key]
        }
        __syncthreads();

        // per-column padding-mask bias (column index is fixed per lane per tile)
        float bias[4];
#pragma unroll
        for (int nt = 0; nt < 4; ++nt)
            bias[nt] = -1e6f * (1.0f - mrow[kb * BLK + nt * 16 + l15]);

        // ---- scores: S[nt] = Q(16x64) * K^T tile (64x16), 2 WMMAs each ----
        v8f s[4];
#pragma unroll
        for (int nt = 0; nt < 4; ++nt) {
            const _Float16* krow = sK + (nt * 16 + l15) * LDSS;
            v16h b0 = ld_frag_lds(krow, c0);
            v16h b1 = ld_frag_lds(krow, 32 + c0);
            v8f z = (v8f)0.0f;
            s[nt] = __builtin_amdgcn_wmma_f32_16x16x32_f16(false, qf0, false, b0,
                                                           (short)0, z, false, false);
            s[nt] = __builtin_amdgcn_wmma_f32_16x16x32_f16(false, qf1, false, b1,
                                                           (short)0, s[nt], false, false);
#pragma unroll
            for (int r = 0; r < 8; ++r) s[nt][r] += bias[nt];
        }

        // ---- online softmax update ----
        float mnew[8], scale[8], rsum[8];
#pragma unroll
        for (int r = 0; r < 8; ++r) {
            float t = s[0][r];
            t = fmaxf(t, s[1][r]); t = fmaxf(t, s[2][r]); t = fmaxf(t, s[3][r]);
#pragma unroll
            for (int off = 1; off < 16; off <<= 1)
                t = fmaxf(t, __shfl_xor(t, off, 16));  // row lives across 16 lanes
            mnew[r]  = fmaxf(mrun[r], t);
            scale[r] = __expf(mrun[r] - mnew[r]);
            rsum[r]  = 0.0f;
        }
        // exponentiate, write P (f16) to this wave's LDS buffer in C/D layout
#pragma unroll
        for (int nt = 0; nt < 4; ++nt) {
#pragma unroll
            for (int r = 0; r < 8; ++r) {
                float p = __expf(s[nt][r] - mnew[r]);
                rsum[r] += p;
                sP[(wave * 16 + ((lane & 16) ? 8 : 0) + r) * LDSS + nt * 16 + l15] =
                    (_Float16)p;
            }
        }
#pragma unroll
        for (int r = 0; r < 8; ++r) {
#pragma unroll
            for (int off = 1; off < 16; off <<= 1)
                rsum[r] += __shfl_xor(rsum[r], off, 16);
            lrun[r] = lrun[r] * scale[r] + rsum[r];
            mrun[r] = mnew[r];
#pragma unroll
            for (int ot = 0; ot < 4; ++ot) acc[ot][r] *= scale[r];
        }

        // wave-local LDS RAW: make P stores visible before A-fragment reloads
        asm volatile("s_wait_dscnt 0" ::: "memory");

        // ---- O += P(16x64) * V(64x64): reload P as A-fragments, V^T as B ----
        v16h p0 = ld_frag_lds(pbase, c0);
        v16h p1 = ld_frag_lds(pbase, 32 + c0);
#pragma unroll
        for (int ot = 0; ot < 4; ++ot) {
            const _Float16* vrow = sV + (ot * 16 + l15) * LDSS;
            v16h vb0 = ld_frag_lds(vrow, c0);
            v16h vb1 = ld_frag_lds(vrow, 32 + c0);
            acc[ot] = __builtin_amdgcn_wmma_f32_16x16x32_f16(false, p0, false, vb0,
                                                             (short)0, acc[ot], false, false);
            acc[ot] = __builtin_amdgcn_wmma_f32_16x16x32_f16(false, p1, false, vb1,
                                                             (short)0, acc[ot], false, false);
        }
    }

    // ---- finalize: O / l, store f32 ----
#pragma unroll
    for (int r = 0; r < 8; ++r) {
        const int rowg = qb * BLK + wave * 16 + ((lane & 16) ? 8 : 0) + r;
        const float inv = 1.0f / lrun[r];
        float* orow = Obh + (long)rowg * DIM;
#pragma unroll
        for (int ot = 0; ot < 4; ++ot)
            orow[ot * 16 + l15] = acc[ot][r] * inv;
    }
}

extern "C" void kernel_launch(void* const* d_in, const int* in_sizes, int n_in,
                              void* d_out, int out_size, void* d_ws, size_t ws_size,
                              hipStream_t stream) {
    const float* q    = (const float*)d_in[0];
    const float* k    = (const float*)d_in[1];
    const float* v    = (const float*)d_in[2];
    const float* mask = (const float*)d_in[3];
    const int*   mat  = (const int*)d_in[4];
    float*       out  = (float*)d_out;
    (void)in_sizes; (void)n_in; (void)out_size; (void)d_ws; (void)ws_size;

    dim3 grid(NB, HEADS, BATCH);   // (32, 8, 2) = 512 workgroups
    dim3 block(128);               // 4 waves, one 16-row Q strip each
    block_mask_attn_kernel<<<grid, block, 0, stream>>>(q, k, v, mask, mat, out);
}
